// GCN_25546465476774
// MI455X (gfx1250) — matrix operand order repaired and checked by reading, not compiled, not run on hardware
//
#include <hip/hip_runtime.h>
#include <hip/hip_bf16.h>

// ---- problem constants (match reference) ----
#define CB 32
#define CN 512
#define CE 8192
#define CD 256
#define CT (CB * CN)              // 16384 rows
#define CTD ((size_t)CT * CD)     // 4,194,304 elems
#define NTILES_N 32               // 512/16
#define PAIRS ((NTILES_N * (NTILES_N + 1)) / 2)   // 528 upper-tri tile pairs
#define ADJ_WORDS (CN * CN / 32)  // 8192 u32 per graph

typedef __attribute__((ext_vector_type(16))) __bf16 v16bf;
typedef __attribute__((ext_vector_type(8)))  __bf16 v8bf;
typedef __attribute__((ext_vector_type(8)))  float  v8f;

__device__ __forceinline__ v16bf cat16(v8bf lo, v8bf hi) {
  union { v16bf v; v8bf h[2]; } u;
  u.h[0] = lo; u.h[1] = hi;
  return u.v;
}

__device__ __forceinline__ float softplus_f(float x) {
  return fmaxf(x, 0.f) + __logf(1.f + __expf(-fabsf(x)));
}

__device__ __forceinline__ float wave_sum(float v) {
  #pragma unroll
  for (int off = 16; off > 0; off >>= 1) v += __shfl_xor(v, off, 32);
  return v;
}

// ------------------------------------------------------------------ utilities
__global__ void zero_f32(float* p, size_t n) {
  size_t i = (size_t)blockIdx.x * blockDim.x + threadIdx.x;
  size_t stride = (size_t)gridDim.x * blockDim.x;
  for (; i < n; i += stride) p[i] = 0.f;
}

// degrees: one thread per edge
__global__ void degrees_kernel(const int* __restrict__ src, const int* __restrict__ dst,
                               float* degO, float* degI) {
  int i = blockIdx.x * blockDim.x + threadIdx.x;
  if (i >= CB * CE) return;
  int b = i / CE;
  atomicAdd(&degO[b * CN + src[i]], 1.f);
  atomicAdd(&degI[b * CN + dst[i]], 1.f);
}

// weight prep: Wt[n][k] = (bf16) W[k][n]
__global__ void wprep_kernel(const float* __restrict__ W, __bf16* __restrict__ Wt) {
  int i = blockIdx.x * blockDim.x + threadIdx.x;
  if (i >= CD * CD) return;
  int k = i / CD, n = i % CD;
  Wt[(size_t)n * CD + k] = (__bf16)W[i];
}

// scatter: agg[dst] += x[src] * rsqrt(max(deg_out[src],1)); float4 per thread
__global__ void scatter_kernel(const float* __restrict__ x,
                               const int* __restrict__ src, const int* __restrict__ dst,
                               const float* __restrict__ degO, float* __restrict__ agg) {
  int i = blockIdx.x * blockDim.x + threadIdx.x;   // CB*CE*64 threads
  if (i >= CB * CE * (CD / 4)) return;
  int c4 = i & 63;
  int eg = i >> 6;
  int b  = eg / CE;
  int s  = src[eg], d = dst[eg];
  float sc = rsqrtf(fmaxf(degO[b * CN + s], 1.f));
  const float4 v = *(const float4*)(x + ((size_t)(b * CN + s)) * CD + c4 * 4);
  float* o = agg + ((size_t)(b * CN + d)) * CD + c4 * 4;
  atomicAdd(o + 0, v.x * sc);
  atomicAdd(o + 1, v.y * sc);
  atomicAdd(o + 2, v.z * sc);
  atomicAdd(o + 3, v.w * sc);
}

// scale by rsqrt(deg_in) and convert to bf16
__global__ void scale_cvt_kernel(const float* __restrict__ agg,
                                 const float* __restrict__ degI,
                                 __bf16* __restrict__ outbf) {
  int i = blockIdx.x * blockDim.x + threadIdx.x;   // CTD/4 threads
  if ((size_t)i >= CTD / 4) return;
  size_t idx = (size_t)i * 4;
  int row = (int)(idx / CD);
  float sc = rsqrtf(fmaxf(degI[row], 1.f));
  float4 v = *(const float4*)(agg + idx);
  outbf[idx + 0] = (__bf16)(v.x * sc);
  outbf[idx + 1] = (__bf16)(v.y * sc);
  outbf[idx + 2] = (__bf16)(v.z * sc);
  outbf[idx + 3] = (__bf16)(v.w * sc);
}

// -------------------------------------------------------------- WMMA GEMM
// C[M,256] = A_bf16[M,256] @ W (Wt is W transposed, bf16) + bias
// one 16x16 tile per wave, K-loop of 8 x v_wmma_f32_16x16x32_bf16
__global__ __launch_bounds__(256)
void gemm_kernel(const __bf16* __restrict__ A, const __bf16* __restrict__ Wt,
                 const float* __restrict__ bias, int relu,
                 float* outF, float* outF2, __bf16* outBF,
                 const float* __restrict__ sqref, float* sqacc) {
  int wave = threadIdx.x >> 5;
  int lane = threadIdx.x & 31;
  int tile = blockIdx.x * 8 + wave;
  int mt = tile / (CD / 16);
  int nt = tile % (CD / 16);
  if (mt * 16 >= CT) return;
  int half = lane >> 4, r = lane & 15;

  const __bf16* arow = A  + (size_t)(mt * 16 + r) * CD;
  const __bf16* brow = Wt + (size_t)(nt * 16 + r) * CD;
  v8f acc = {};
  #pragma unroll
  for (int k0 = 0; k0 < CD; k0 += 32) {
    v8bf alo = *(const v8bf*)(arow + k0 + half * 8);
    v8bf ahi = *(const v8bf*)(arow + k0 + 16 + half * 8);
    v8bf blo = *(const v8bf*)(brow + k0 + half * 8);
    v8bf bhi = *(const v8bf*)(brow + k0 + 16 + half * 8);
    acc = __builtin_amdgcn_wmma_f32_16x16x32_bf16(
        false, cat16(alo, ahi), false, cat16(blo, bhi),
        (short)0, acc, false, false);
  }

  int col = nt * 16 + r;
  float bv = bias ? bias[col] : 0.f;
  float lsum = 0.f;
  #pragma unroll
  for (int j = 0; j < 8; j++) {
    int row = mt * 16 + half * 8 + j;
    float v = acc[j] + bv;
    if (relu) v = fmaxf(v, 0.f);
    size_t o = (size_t)row * CD + col;
    if (outF)  outF[o]  = v;
    if (outF2) outF2[o] = v;
    if (outBF) outBF[o] = (__bf16)v;
    if (sqref) { float dlt = sqref[o] - v; lsum += dlt * dlt; }
  }
  if (sqref) {
    lsum = wave_sum(lsum);
    if (lane == 0) atomicAdd(sqacc, lsum);
  }
}

// layernorm in place, one wave per 256-float row
__global__ __launch_bounds__(256)
void layernorm_kernel(float* __restrict__ x, const float* __restrict__ g,
                      const float* __restrict__ bb) {
  int wave = threadIdx.x >> 5, lane = threadIdx.x & 31;
  int row = blockIdx.x * 8 + wave;
  if (row >= CT) return;
  float* p = x + (size_t)row * CD;
  float vals[8];
  float4 v0 = *(const float4*)(p + lane * 8);
  float4 v1 = *(const float4*)(p + lane * 8 + 4);
  vals[0] = v0.x; vals[1] = v0.y; vals[2] = v0.z; vals[3] = v0.w;
  vals[4] = v1.x; vals[5] = v1.y; vals[6] = v1.z; vals[7] = v1.w;
  float s = 0.f;
  #pragma unroll
  for (int i = 0; i < 8; i++) s += vals[i];
  float mu = wave_sum(s) * (1.f / CD);
  float q = 0.f;
  #pragma unroll
  for (int i = 0; i < 8; i++) { float d = vals[i] - mu; q += d * d; }
  float rstd = rsqrtf(wave_sum(q) * (1.f / CD) + 1e-5f);
  #pragma unroll
  for (int i = 0; i < 8; i++) {
    int c = lane * 8 + i;
    p[c] = (vals[i] - mu) * rstd * g[c] + bb[c];
  }
}

// adjacency bitmask (set semantics, only s<d bits are ever read by the triu loss)
__global__ void adj_build_kernel(const int* __restrict__ src, const int* __restrict__ dst,
                                 unsigned* __restrict__ bits) {
  int i = blockIdx.x * blockDim.x + threadIdx.x;
  if (i >= CB * CE) return;
  int b = i / CE;
  int s = src[i], d = dst[i];
  if (s < d) {
    unsigned idx = (unsigned)(s * CN + d);
    atomicOr(&bits[(size_t)b * ADJ_WORDS + (idx >> 5)], 1u << (idx & 31));
  }
}

// per-graph popcount -> pos_w
__global__ __launch_bounds__(256)
void adj_count_kernel(const unsigned* __restrict__ bits, float* __restrict__ posw) {
  __shared__ int sh[256];
  int b = blockIdx.x;
  const unsigned* p = bits + (size_t)b * ADJ_WORDS;
  int cnt = 0;
  for (int w = threadIdx.x; w < ADJ_WORDS; w += 256) cnt += __popc(p[w]);
  sh[threadIdx.x] = cnt;
  __syncthreads();
  for (int st = 128; st > 0; st >>= 1) {
    if (threadIdx.x < st) sh[threadIdx.x] += sh[threadIdx.x + st];
    __syncthreads();
  }
  if (threadIdx.x == 0) {
    float ne = (float)sh[0];
    posw[b] = ((float)CN * (float)CN * 0.5f - ne) / (ne + 1e-6f);
  }
}

// qe @ qe^T over upper-triangular 16x16 tile pairs + fused masked softplus BCE
__global__ __launch_bounds__(256)
void edge_loss_kernel(const __bf16* __restrict__ qebf,
                      const unsigned* __restrict__ bits,
                      const float* __restrict__ posw,
                      float* __restrict__ edge_loss) {
  int wave = threadIdx.x >> 5, lane = threadIdx.x & 31;
  int job = blockIdx.x * 8 + wave;
  int b = job / PAIRS;
  int p = job % PAIRS;
  if (b >= CB) return;
  int nt = 0;
  while (p >= NTILES_N - nt) { p -= NTILES_N - nt; nt++; }
  int mt = nt + p;
  int half = lane >> 4, r = lane & 15;

  const __bf16* arow = qebf + (size_t)(b * CN + nt * 16 + r) * CD;
  const __bf16* brow = qebf + (size_t)(b * CN + mt * 16 + r) * CD;
  v8f acc = {};
  #pragma unroll
  for (int k0 = 0; k0 < CD; k0 += 32) {
    v8bf alo = *(const v8bf*)(arow + k0 + half * 8);
    v8bf ahi = *(const v8bf*)(arow + k0 + 16 + half * 8);
    v8bf blo = *(const v8bf*)(brow + k0 + half * 8);
    v8bf bhi = *(const v8bf*)(brow + k0 + 16 + half * 8);
    acc = __builtin_amdgcn_wmma_f32_16x16x32_bf16(
        false, cat16(alo, ahi), false, cat16(blo, bhi),
        (short)0, acc, false, false);
  }

  float pw = posw[b];
  const unsigned* bp = bits + (size_t)b * ADJ_WORDS;
  int m = mt * 16 + r;
  float lsum = 0.f;
  #pragma unroll
  for (int j = 0; j < 8; j++) {
    int n = nt * 16 + half * 8 + j;
    if (n < m) {
      float x = acc[j];
      unsigned idx = (unsigned)(n * CN + m);
      bool y = (bp[idx >> 5] >> (idx & 31)) & 1u;
      lsum += y ? pw * softplus_f(-x) : softplus_f(x);
    }
  }
  lsum = wave_sum(lsum);
  if (lane == 0) atomicAdd(&edge_loss[b], lsum);
}

__global__ void finalize_kernel(const float* __restrict__ sqacc,
                                const float* __restrict__ edge_loss,
                                float* __restrict__ out_loss) {
  if (threadIdx.x == 0 && blockIdx.x == 0) {
    float feat_rec = sqacc[0] / (float)CTD;          // LAMB_NODE = 1
    float es = 0.f;
    for (int b = 0; b < CB; b++) es += edge_loss[b];
    float n_mask = (float)(CN * (CN - 1) / 2);       // 130816
    float edge_rec = (es / n_mask) / (float)CB;      // LAMB_EDGE = 1
    out_loss[0] = feat_rec + 100.0f * edge_rec;      // ALPHA = 100
  }
}

// ----------------------------------------------------------------- launcher
extern "C" void kernel_launch(void* const* d_in, const int* in_sizes, int n_in,
                              void* d_out, int out_size, void* d_ws, size_t ws_size,
                              hipStream_t stream) {
  (void)in_sizes; (void)n_in; (void)out_size; (void)ws_size;
  const float* feats = (const float*)d_in[0];
  const int*   src   = (const int*)d_in[1];
  const int*   dst   = (const int*)d_in[2];
  const float* Wg1   = (const float*)d_in[3];
  const float* bg1   = (const float*)d_in[4];
  const float* Wg2   = (const float*)d_in[5];
  const float* bg2   = (const float*)d_in[6];
  const float* ln_g  = (const float*)d_in[7];
  const float* ln_b  = (const float*)d_in[8];
  const float* W1    = (const float*)d_in[9];
  const float* b1    = (const float*)d_in[10];
  const float* W2    = (const float*)d_in[11];
  const float* b2    = (const float*)d_in[12];

  float* out   = (float*)d_out;
  float* h1    = out;
  float* h2a   = out + CTD;
  float* qe    = out + 2 * CTD;
  float* h2b   = out + 3 * CTD;
  float* lossp = out + 4 * CTD;

  // workspace layout (256B-aligned regions; zeroed prefix first)
  char* ws = (char*)d_ws;
  size_t off = 0;
  auto alloc = [&](size_t bytes) -> void* {
    void* p = ws + off;
    off = (off + bytes + 255) & ~(size_t)255;
    return p;
  };
  float*    degO      = (float*)alloc((size_t)CT * 4);
  float*    degI      = (float*)alloc((size_t)CT * 4);
  float*    edgeLoss  = (float*)alloc((size_t)CB * 4);
  float*    sqacc     = (float*)alloc(64);
  unsigned* adjbits   = (unsigned*)alloc((size_t)CB * ADJ_WORDS * 4);
  size_t zeroBytes = off;                       // everything above starts at 0
  float*    posw      = (float*)alloc((size_t)CB * 4);
  float*    agg       = (float*)alloc(CTD * 4);
  __bf16*   Abf       = (__bf16*)alloc(CTD * 2);
  __bf16*   h2bf      = (__bf16*)alloc(CTD * 2);
  __bf16*   qebf      = (__bf16*)alloc(CTD * 2);
  __bf16*   Wg1t      = (__bf16*)alloc((size_t)CD * CD * 2);
  __bf16*   Wg2t      = (__bf16*)alloc((size_t)CD * CD * 2);
  __bf16*   W1t       = (__bf16*)alloc((size_t)CD * CD * 2);
  __bf16*   W2t       = (__bf16*)alloc((size_t)CD * CD * 2);

  const int TPB = 256;
  // 0) zero accumulators / degrees / adjacency bitmask
  zero_f32<<<4096, TPB, 0, stream>>>((float*)ws, zeroBytes / 4);
  // 1) degrees + weight prep + adjacency (independent)
  degrees_kernel<<<(CB * CE) / TPB, TPB, 0, stream>>>(src, dst, degO, degI);
  wprep_kernel<<<(CD * CD) / TPB, TPB, 0, stream>>>(Wg1, Wg1t);
  wprep_kernel<<<(CD * CD) / TPB, TPB, 0, stream>>>(Wg2, Wg2t);
  wprep_kernel<<<(CD * CD) / TPB, TPB, 0, stream>>>(W1, W1t);
  wprep_kernel<<<(CD * CD) / TPB, TPB, 0, stream>>>(W2, W2t);
  adj_build_kernel<<<(CB * CE) / TPB, TPB, 0, stream>>>(src, dst, adjbits);
  adj_count_kernel<<<CB, TPB, 0, stream>>>(adjbits, posw);

  // 2) GraphConv1: scatter -> scale/convert -> WMMA GEMM(+bias,relu) -> LN
  zero_f32<<<4096, TPB, 0, stream>>>(agg, CTD);
  scatter_kernel<<<(CB * CE * (CD / 4)) / TPB, TPB, 0, stream>>>(feats, src, dst, degO, agg);
  scale_cvt_kernel<<<(int)(CTD / 4 / TPB), TPB, 0, stream>>>(agg, degI, Abf);
  gemm_kernel<<<(CT / 16) * (CD / 16) / 8, TPB, 0, stream>>>(
      Abf, Wg1t, bg1, /*relu=*/1, h1, nullptr, nullptr, nullptr, nullptr);
  layernorm_kernel<<<CT / 8, TPB, 0, stream>>>(h1, ln_g, ln_b);

  // 3) GraphConv2 on h1 -> h2 (written to both output slots + bf16 copy)
  zero_f32<<<4096, TPB, 0, stream>>>(agg, CTD);
  scatter_kernel<<<(CB * CE * (CD / 4)) / TPB, TPB, 0, stream>>>(h1, src, dst, degO, agg);
  scale_cvt_kernel<<<(int)(CTD / 4 / TPB), TPB, 0, stream>>>(agg, degI, Abf);
  gemm_kernel<<<(CT / 16) * (CD / 16) / 8, TPB, 0, stream>>>(
      Abf, Wg2t, bg2, /*relu=*/1, h2a, h2b, h2bf, nullptr, nullptr);

  // 4) decoders: qe (stored, + bf16 for einsum); qn fused into sum((h2-qn)^2)
  gemm_kernel<<<(CT / 16) * (CD / 16) / 8, TPB, 0, stream>>>(
      h2bf, W1t, b1, /*relu=*/0, qe, nullptr, qebf, nullptr, nullptr);
  gemm_kernel<<<(CT / 16) * (CD / 16) / 8, TPB, 0, stream>>>(
      h2bf, W2t, b2, /*relu=*/0, nullptr, nullptr, nullptr, h2a, sqacc);

  // 5) adjacency reconstruction loss: WMMA qe@qe^T over upper-tri tiles
  edge_loss_kernel<<<(CB * PAIRS) / 8, TPB, 0, stream>>>(qebf, adjbits, posw, edgeLoss);

  // 6) combine
  finalize_kernel<<<1, 32, 0, stream>>>(sqacc, edgeLoss, lossp);
}
